// LocalisedSpectrumEncoder_81698867905051
// MI455X (gfx1250) — matrix operand order repaired and checked by reading, not compiled, not run on hardware
//
#include <hip/hip_runtime.h>
#include <math.h>

#define BATCH 16
#define NPK 150
#define NTOK 151
#define DIM 256
#define NH 8
#define HD 32
#define FFN_DIM 1024
#define ROWS_TOT (BATCH * NTOK)     // 2416
#define ROWS_PAD 2432               // next multiple of 32
#define PEROWS 20200
#define PEROWS_PAD 20224            // next multiple of 32

typedef __attribute__((ext_vector_type(16))) _Float16 v16h;
typedef __attribute__((ext_vector_type(8)))  _Float16 v8h;
typedef __attribute__((ext_vector_type(8)))  float    v8f;

// ---------------------------------------------------------------------------
// Embedding: x[b,0,:] = latent ; x[b,1+p,:] = sinusoid(mz) + inten*int_w
// Also emits masses (B*N) and pad mask (B*N).
// ---------------------------------------------------------------------------
__global__ void k_embed(const float* __restrict__ spectra,
                        const float* __restrict__ latent,
                        const float* __restrict__ int_w,
                        float* __restrict__ x,
                        float* __restrict__ masses,
                        int* __restrict__ padf)
{
    int row = blockIdx.x;            // b*NTOK + n
    int t   = threadIdx.x;           // 0..255
    int n   = row % NTOK;
    int b   = row / NTOK;
    if (n == 0) {
        x[row * DIM + t] = latent[t];
        if (t == 0) { masses[row] = 0.0f; padf[row] = 0; }
        return;
    }
    int p = n - 1;
    float mz  = spectra[(b * NPK + p) * 2 + 0];
    float ity = spectra[(b * NPK + p) * 2 + 1];
    const float basec = 0.001f / (2.0f * 3.14159265358979323846f);
    float val;
    if (t < 128) {
        float term = basec * __powf(1.0e7f, (float)t * (1.0f / 127.0f));
        val = __sinf(mz / term);
    } else {
        float term = basec * __powf(1.0e7f, (float)(t - 128) * (1.0f / 127.0f));
        val = __cosf(mz / term);
    }
    val += ity * int_w[t];
    x[row * DIM + t] = val;
    if (t == 0) { masses[row] = mz; padf[row] = ((mz + ity) == 0.0f) ? 1 : 0; }
}

// q_in = x + pe[idx(mass=0)] ; idx = (0 + 1 + 100)/0.01 = 10100
__global__ void k_add_pe0(const float* __restrict__ x,
                          const float* __restrict__ pe,
                          float* __restrict__ out)
{
    int row = blockIdx.x, t = threadIdx.x;
    out[row * DIM + t] = x[row * DIM + t] + pe[10100 * DIM + t];
}

// ---------------------------------------------------------------------------
// WMMA GEMM: C[Mt,Nn] = A[Msrc,K] @ W[Nn,K]^T (+bias, +relu).
// Each wave32 owns a 32x64 output strip and a private LDS slice: 2 A
// fragments x 4 B fragments -> 8 v_wmma_f32_16x16x32_f16 per 32-deep k-step.
// Staging is lane==row: each lane reads its own row's 128B chunk through ONE
// base pointer with immediate offsets (clause-friendly, no address pressure),
// converts to f16 and writes v8h ds_store_b128s, one array at a time.
// No workgroup barriers: same-wave LDS ops retire in order (DScnt in-order),
// so a wave-local s_wait_dscnt drains staging stores before fragment loads.
// __launch_bounds__(256,1) unlocks the full VGPR budget (no spills): live set
// is 64 acc + 32 staging + ~48 fragment + addresses.
// Mt is a multiple of 32, C has Mt rows (caller pads workspace); A-source
// rows clamp to Msrc-1 -> zero guards anywhere.
// ---------------------------------------------------------------------------
template <bool HAS_BIAS, bool DO_RELU>
__global__ __launch_bounds__(256, 1)
void k_wmma_gemm(const float* __restrict__ A,
                 const float* __restrict__ W,
                 const float* __restrict__ bias,
                 float* __restrict__ C,
                 int Mt, int Msrc, int Nn, int K)
{
    __shared__ __align__(16) _Float16 sA[8][32 * 32];   // [m][k] per wave
    __shared__ __align__(16) _Float16 sB[8][64 * 32];   // [n][k] per wave

    const int wave   = threadIdx.x >> 5;
    const int lane   = threadIdx.x & 31;
    const int half   = lane >> 4;
    const int rn     = lane & 15;
    const int tilesN = Nn >> 6;                         // 64-wide N strips
    const int tile   = blockIdx.x * 8 + wave;
    if (tile >= (Mt >> 5) * tilesN) return;             // wave-uniform
    const int tm = (tile / tilesN) << 5;
    const int tn = (tile % tilesN) << 6;

    _Float16* __restrict__ la = sA[wave];
    _Float16* __restrict__ lb = sB[wave];

    // lane == row: one base pointer per owned row, k-invariant.
    int garow = tm + lane; if (garow >= Msrc) garow = Msrc - 1;
    const float* pA  = A + (size_t)garow * K;
    const float* pB0 = W + (size_t)(tn + lane) * K;
    const float* pB1 = W + (size_t)(tn + 32 + lane) * K;
    _Float16* qA  = &la[lane * 32];
    _Float16* qB0 = &lb[lane * 32];
    _Float16* qB1 = &lb[(lane + 32) * 32];

    v8f acc[2][4] = {};

    for (int k0 = 0; k0 < K; k0 += 32) {
        // ---- stage one 32-float row chunk per owned row, array by array ----
#pragma unroll
        for (int arr = 0; arr < 3; ++arr) {
            const float* src = (arr == 0) ? pA : (arr == 1) ? pB0 : pB1;
            _Float16*    dst = (arr == 0) ? qA : (arr == 1) ? qB0 : qB1;
            float4 t[8];
#pragma unroll
            for (int it = 0; it < 8; ++it)
                t[it] = *(const float4*)(src + k0 + it * 4);
#pragma unroll
            for (int c = 0; c < 4; ++c) {
                v8h hh = { (_Float16)t[2*c].x,   (_Float16)t[2*c].y,
                           (_Float16)t[2*c].z,   (_Float16)t[2*c].w,
                           (_Float16)t[2*c+1].x, (_Float16)t[2*c+1].y,
                           (_Float16)t[2*c+1].z, (_Float16)t[2*c+1].w };
                *(v8h*)(dst + c * 8) = hh;
            }
        }
        // exact per-lane prefetch of next k-step (speculative: dropped if OOB)
        if (k0 + 32 < K) {
            __builtin_prefetch(pA  + k0 + 32, 0, 3);
            __builtin_prefetch(pB0 + k0 + 32, 0, 3);
            __builtin_prefetch(pB1 + k0 + 32, 0, 3);
        }
        // Wave-local LDS handoff: drain staging stores; block reordering.
        asm volatile("s_wait_dscnt 0x0" ::: "memory");

        // A fragments (16-bit 16x32 layout): a[0..7]=K 8h..8h+7,
        // a[8..15]=K 16+8h..+7 for rows rn and rn+16.
        v16h av0, av1, bv[4];
        {
            const _Float16* pa0 = &la[rn * 32];
            const _Float16* pa1 = &la[(16 + rn) * 32];
            v8h x0 = *(const v8h*)(pa0 + half * 8);
            v8h x1 = *(const v8h*)(pa0 + 16 + half * 8);
            v8h y0 = *(const v8h*)(pa1 + half * 8);
            v8h y1 = *(const v8h*)(pa1 + 16 + half * 8);
#pragma unroll
            for (int i = 0; i < 8; ++i) {
                av0[i] = x0[i]; av0[i + 8] = x1[i];
                av1[i] = y0[i]; av1[i + 8] = y1[i];
            }
        }
#pragma unroll
        for (int t = 0; t < 4; ++t) {                   // B fragments
            const _Float16* pb = &lb[(t * 16 + rn) * 32 + half * 8];
            v8h b0 = *(const v8h*)pb;
            v8h b1 = *(const v8h*)(pb + 16);
#pragma unroll
            for (int i = 0; i < 8; ++i) { bv[t][i] = b0[i]; bv[t][i + 8] = b1[i]; }
        }
#pragma unroll
        for (int t = 0; t < 4; ++t) {
            acc[0][t] = __builtin_amdgcn_wmma_f32_16x16x32_f16(
                false, av0, false, bv[t], (short)0, acc[0][t], false, false);
            acc[1][t] = __builtin_amdgcn_wmma_f32_16x16x32_f16(
                false, av1, false, bv[t], (short)0, acc[1][t], false, false);
        }
    }

    // Epilogue. C layout: VGPR v -> row v + 8*half, col = lane&15 per 16-tile.
    const size_t sN = (size_t)Nn;
    float b0 = 0.0f, b1 = 0.0f, b2 = 0.0f, b3 = 0.0f;
    if (HAS_BIAS) {
        b0 = bias[tn + rn];      b1 = bias[tn + 16 + rn];
        b2 = bias[tn + 32 + rn]; b3 = bias[tn + 48 + rn];
    }
#pragma unroll
    for (int mt = 0; mt < 2; ++mt) {
        float* cp = C + (size_t)(tm + mt * 16 + half * 8) * sN + tn + rn;
#pragma unroll
        for (int v = 0; v < 8; ++v) {
#pragma unroll
            for (int t = 0; t < 4; ++t) {
                float o = acc[mt][t][v] + (t == 0 ? b0 : t == 1 ? b1 : t == 2 ? b2 : b3);
                if (DO_RELU) o = fmaxf(o, 0.0f);
                cp[t * 16] = o;
            }
            cp += sN;
        }
    }
}

// ---------------------------------------------------------------------------
// Flash-style attention. One wave32 per (b,i,h); lane == head-dim (hd=32).
// useRel=1: K/V rows get the gathered projected-PE contribution added
// (K[b,i,j] = xk[b,j] + peK[idx(i,j)]), exploiting linearity of projection.
// ---------------------------------------------------------------------------
__global__ void k_attn(const float* __restrict__ Q,
                       const float* __restrict__ XK,
                       const float* __restrict__ XV,
                       const float* __restrict__ peK,
                       const float* __restrict__ peV,
                       const float* __restrict__ masses,
                       const int* __restrict__ padf,
                       float* __restrict__ Y,
                       int useRel)
{
    int gw   = (blockIdx.x * blockDim.x + threadIdx.x) >> 5;
    int lane = threadIdx.x & 31;
    int h    = gw % NH;
    int bi   = gw / NH;                         // b*NTOK + i
    if (bi >= ROWS_TOT) return;                 // wave-uniform
    int b   = bi / NTOK;
    int off = h * HD + lane;

    float q  = Q[(size_t)bi * DIM + off] * 0.17677669529663688f;  // 1/sqrt(32)
    float mi = masses[bi];
    float mrun = -1.0e30f, denom = 0.0f, acc = 0.0f;
    int base0 = b * NTOK;

    for (int j = 0; j < NTOK; ++j) {
        int rj = base0 + j;
        size_t a = (size_t)rj * DIM + off;
        float kv = XK[a];
        float vv = XV[a];
        if (useRel) {
            float rel = masses[rj] - mi;
            rel = fminf(fmaxf(rel, -100.0f), 100.0f);
            int idx = (int)((rel + 101.0f) / 0.01f);
            size_t pa = (size_t)idx * DIM + off;
            kv += peK[pa];
            vv += peV[pa];
        }
        float s = q * kv;
#pragma unroll
        for (int d = 16; d >= 1; d >>= 1) s += __shfl_xor(s, d, 32);
        if (padf[rj]) s = -1.0e9f;
        float mnew  = fmaxf(mrun, s);
        float alpha = __expf(mrun - mnew);
        float p     = __expf(s - mnew);
        acc   = acc * alpha + p * vv;
        denom = denom * alpha + p;
        mrun  = mnew;
    }
    Y[(size_t)bi * DIM + off] = acc / denom;
}

// ---------------------------------------------------------------------------
// out = LayerNorm(xin + r) * g + b   (one block per row, 256 threads)
// ---------------------------------------------------------------------------
__global__ void k_resid_ln(const float* __restrict__ xin,
                           const float* __restrict__ r,
                           const float* __restrict__ g,
                           const float* __restrict__ bb,
                           float* __restrict__ out)
{
    __shared__ float red[DIM];
    int row = blockIdx.x, t = threadIdx.x;
    float v = xin[(size_t)row * DIM + t] + r[(size_t)row * DIM + t];
    red[t] = v; __syncthreads();
    for (int s = 128; s >= 1; s >>= 1) { if (t < s) red[t] += red[t + s]; __syncthreads(); }
    float mu = red[0] * (1.0f / DIM);
    __syncthreads();
    float d = v - mu;
    red[t] = d * d; __syncthreads();
    for (int s = 128; s >= 1; s >>= 1) { if (t < s) red[t] += red[t + s]; __syncthreads(); }
    float var = red[0] * (1.0f / DIM);
    out[(size_t)row * DIM + t] = d * rsqrtf(var + 1e-5f) * g[t] + bb[t];
}

// ---------------------------------------------------------------------------
extern "C" void kernel_launch(void* const* d_in, const int* in_sizes, int n_in,
                              void* d_out, int out_size, void* d_ws, size_t ws_size,
                              hipStream_t stream)
{
    (void)in_sizes; (void)n_in; (void)out_size; (void)ws_size;
    const float* spectra = (const float*)d_in[0];
    const float* pe      = (const float*)d_in[1];
    const float* latent  = (const float*)d_in[2];
    const float* int_w   = (const float*)d_in[3];
    const float* in_w    = (const float*)d_in[4];
    const float* in_b    = (const float*)d_in[5];
    const float* out_w   = (const float*)d_in[6];
    const float* out_b   = (const float*)d_in[7];
    const float* l1_w    = (const float*)d_in[8];
    const float* l1_b    = (const float*)d_in[9];
    const float* l2_w    = (const float*)d_in[10];
    const float* l2_b    = (const float*)d_in[11];
    const float* n1_g    = (const float*)d_in[12];
    const float* n1_b    = (const float*)d_in[13];
    const float* n2_g    = (const float*)d_in[14];
    const float* n2_b    = (const float*)d_in[15];

    char* wp = (char*)d_ws;
    auto alloc = [&](size_t bytes) -> void* {
        void* p = wp; wp += (bytes + 255) & ~(size_t)255; return p;
    };
    // GEMM outputs get ROWS_PAD rows so stores never need guards; padded rows
    // are never consumed (A-reads clamp to Msrc, attention/LN index < ROWS_TOT).
    float* x      = (float*)alloc((size_t)ROWS_PAD * DIM * 4);
    float* tmp    = (float*)alloc((size_t)ROWS_PAD * DIM * 4);
    float* q      = (float*)alloc((size_t)ROWS_PAD * DIM * 4);
    float* xk     = (float*)alloc((size_t)ROWS_PAD * DIM * 4);
    float* xv     = (float*)alloc((size_t)ROWS_PAD * DIM * 4);
    float* y      = (float*)alloc((size_t)ROWS_PAD * DIM * 4);
    float* hbuf   = (float*)alloc((size_t)ROWS_PAD * FFN_DIM * 4);
    float* pek    = (float*)alloc((size_t)PEROWS_PAD * DIM * 4);
    float* pev    = (float*)alloc((size_t)PEROWS_PAD * DIM * 4);
    float* masses = (float*)alloc((size_t)ROWS_TOT * 4);
    int*   padf   = (int*)alloc((size_t)ROWS_TOT * 4);

    k_embed<<<ROWS_TOT, DIM, 0, stream>>>(spectra, latent, int_w, x, masses, padf);

    auto gemm = [&](const float* A, const float* W, const float* bias, float* C,
                    int Mt, int Msrc, int Nn, int K, bool relu) {
        int tiles  = (Mt / 32) * (Nn / 64);
        int blocks = (tiles + 7) / 8;
        if (bias) {
            if (relu) k_wmma_gemm<true, true><<<blocks, 256, 0, stream>>>(A, W, bias, C, Mt, Msrc, Nn, K);
            else      k_wmma_gemm<true, false><<<blocks, 256, 0, stream>>>(A, W, bias, C, Mt, Msrc, Nn, K);
        } else      k_wmma_gemm<false, false><<<blocks, 256, 0, stream>>>(A, W, nullptr, C, Mt, Msrc, Nn, K);
    };

    // Projected-PE tables for layer-0 relative attention (linearity trick):
    // peK = pe @ wk0^T, peV = pe @ wv0^T  (biases live in xk/xv).
    const float* wk0 = in_w + (size_t)DIM * DIM;
    const float* wv0 = in_w + (size_t)2 * DIM * DIM;
    gemm(pe, wk0, nullptr, pek, PEROWS_PAD, PEROWS, DIM, DIM, false);
    gemm(pe, wv0, nullptr, pev, PEROWS_PAD, PEROWS, DIM, DIM, false);

    for (int l = 0; l < 2; ++l) {
        const float* wq = in_w + (size_t)l * 3 * DIM * DIM;
        const float* wk = wq + (size_t)DIM * DIM;
        const float* wv = wq + (size_t)2 * DIM * DIM;
        const float* bq = in_b + (size_t)l * 3 * DIM;
        const float* bk = bq + DIM;
        const float* bv = bq + 2 * DIM;

        const float* qin = x;
        if (l == 0) { k_add_pe0<<<ROWS_TOT, DIM, 0, stream>>>(x, pe, tmp); qin = tmp; }
        gemm(qin, wq, bq, q,  ROWS_PAD, ROWS_TOT, DIM, DIM, false);
        gemm(x,   wk, bk, xk, ROWS_PAD, ROWS_TOT, DIM, DIM, false);
        gemm(x,   wv, bv, xv, ROWS_PAD, ROWS_TOT, DIM, DIM, false);

        int waves = ROWS_TOT * NH;                   // 19328 waves
        k_attn<<<(waves + 7) / 8, 256, 0, stream>>>(q, xk, xv, pek, pev,
                                                    masses, padf, y, l == 0 ? 1 : 0);

        gemm(y, out_w + (size_t)l * DIM * DIM, out_b + (size_t)l * DIM, tmp,
             ROWS_PAD, ROWS_TOT, DIM, DIM, false);
        k_resid_ln<<<ROWS_TOT, DIM, 0, stream>>>(x, tmp, n1_g + (size_t)l * DIM,
                                                 n1_b + (size_t)l * DIM, x);

        gemm(x, l1_w + (size_t)l * FFN_DIM * DIM, l1_b + (size_t)l * FFN_DIM,
             hbuf, ROWS_PAD, ROWS_TOT, FFN_DIM, DIM, true);
        gemm(hbuf, l2_w + (size_t)l * DIM * FFN_DIM, l2_b + (size_t)l * DIM,
             tmp, ROWS_PAD, ROWS_TOT, DIM, FFN_DIM, false);

        float* dst = (l == 1) ? (float*)d_out : x;
        k_resid_ln<<<ROWS_TOT, DIM, 0, stream>>>(x, tmp, n2_g + (size_t)l * DIM,
                                                 n2_b + (size_t)l * DIM, dst);
    }
}
